// MultiHeadedAttention_88210038325763
// MI455X (gfx1250) — compile-verified, matrix operand order checked by reading, not run on hardware
//
#include <hip/hip_runtime.h>
#include <hip/hip_bf16.h>

#define BB   4
#define DD   256
#define NN   4096
#define HH   4
#define DIMD 64        // per-head dim
#define DBB  32        // bottleneck channels
#define BC   64        // keys per inner block

// softmax scale (1/sqrt(64)) folded together with log2(e) into the Q
// projection, so attention scores are produced directly in the log2 domain
// and softmax uses bare v_exp_f32 (exp2) with no per-element multiplies.
#define QSCALE 0.18033688011112042f   // 0.125 * log2(e)

typedef _Float16 half16_t __attribute__((ext_vector_type(16)));
typedef _Float16 half8_t  __attribute__((ext_vector_type(8)));
typedef float    float8_t __attribute__((ext_vector_type(8)));

// Pack two contiguous 8-half chunks into a WMMA 16-half operand register set.
static __device__ inline half16_t pack16(const _Float16* p0, const _Float16* p1) {
    half8_t lo = *(const half8_t*)p0;
    half8_t hi = *(const half8_t*)p1;
    half16_t r;
#pragma unroll
    for (int i = 0; i < 8; ++i) { r[i] = lo[i]; r[8 + i] = hi[i]; }
    return r;
}

static __device__ inline float8_t wmma_f16(half16_t a, half16_t b, float8_t c) {
    return __builtin_amdgcn_wmma_f32_16x16x32_f16(false, a, false, b, (short)0, c, false, false);
}

// CDNA5 async global->LDS copy of 16B per lane (GLOBAL_LOAD_ASYNC_TO_LDS_B128,
// tracked by ASYNCcnt). lds_off = wave-relative LDS byte address (low 32 bits
// of the generic __shared__ pointer per the aperture mapping rules).
static __device__ inline void async_copy_b128(const void* gptr, uint32_t lds_off) {
    asm volatile("global_load_async_to_lds_b128 %0, %1, off"
                 :: "v"(lds_off), "v"(gptr) : "memory");
}

// ---------------------------------------------------------------------------
// Bottleneck 1x1 conv projection: out = scale*(W2 @ (W1 @ x + b1) + b2), f16.
// mode 0: out[(bh)*N + n][DIM] layout  (q and k: row = token, contiguous d)
// mode 2: out[(bh)*DIM + d][N] layout  (v: row = d, contiguous tokens)
// ---------------------------------------------------------------------------
__global__ __launch_bounds__(128) void proj_kernel(
    const float* __restrict__ x,  const float* __restrict__ W1,
    const float* __restrict__ b1, const float* __restrict__ W2,
    const float* __restrict__ b2, _Float16* __restrict__ out, int mode,
    float out_scale)
{
    __shared__ float W1s[DBB * DD];
    __shared__ float W2s[DD * DBB];
    for (int i = threadIdx.x; i < DBB * DD; i += 128) {
        W1s[i] = W1[i];
        W2s[i] = W2[i];
    }
    __syncthreads();

    const int blk = blockIdx.x;               // B * (N/128) blocks
    const int b   = blk / (NN / 128);
    const int n   = (blk % (NN / 128)) * 128 + threadIdx.x;

    float hacc[DBB];
#pragma unroll
    for (int i = 0; i < DBB; ++i) hacc[i] = b1[i];

    const float* xb = x + (size_t)b * DD * NN + n;
    for (int ci = 0; ci < DD; ++ci) {
        float xv = xb[(size_t)ci * NN];
#pragma unroll
        for (int co = 0; co < DBB; ++co)
            hacc[co] = fmaf(W1s[co * DD + ci], xv, hacc[co]);
    }

    for (int co = 0; co < DD; ++co) {
        float oacc = b2[co];
#pragma unroll
        for (int i = 0; i < DBB; ++i)
            oacc = fmaf(W2s[co * DBB + i], hacc[i], oacc);
        oacc *= out_scale;
        const int hh = co & (HH - 1);     // head  = channel % 4
        const int d  = co >> 2;           // d     = channel / 4
        size_t idx;
        if (mode == 2) idx = (((size_t)(b * HH + hh)) * DIMD + d) * NN + n;
        else           idx = (((size_t)(b * HH + hh)) * NN + n) * DIMD + d;
        out[idx] = (_Float16)oacc;
    }
}

// ---------------------------------------------------------------------------
// Fused flash attention, WMMA f16 -> f32, async double-buffered K/V staging.
// One block = 4 waves; each wave owns 16 query rows (64 per workgroup).
// Computes S^T = K Q^T so the softmax (key) axis lies along VGPRs + a single
// lane^16 exchange; then O += V P^T (the C-tile layout of S^T maps exactly
// onto the f16 B-operand packing of the PV WMMA -- pure in-lane conversion).
// Scores arrive pre-scaled into the log2 domain (QSCALE folded into Q), so
// the softmax inner loop is subtract + v_exp_f32 only.
// 16 v_wmma per 64-key block; next block streams in via ASYNCcnt.
// ---------------------------------------------------------------------------
__global__ __launch_bounds__(128) void attn_kernel(
    const _Float16* __restrict__ q,   // [B*H][N][DIM]  (pre-scaled by QSCALE)
    const _Float16* __restrict__ k,   // [B*H][N][DIM]
    const _Float16* __restrict__ v,   // [B*H][DIM][N]
    float* __restrict__ xout)         // [B][D][N]
{
    __shared__ _Float16 Ks[2][BC * DIMD];   // 64 keys x 64 d   (8 KB x2)
    __shared__ _Float16 Vs[2][DIMD * BC];   // 64 d x 64 keys   (8 KB x2)

    const int bh   = blockIdx.y;
    const int b    = bh / HH;
    const int h    = bh % HH;
    const int wave = threadIdx.x >> 5;
    const int lane = threadIdx.x & 31;
    const int col  = lane & 15;          // query column / A-row selector
    const int hi   = lane >> 4;          // lane half selector

    const int qbase = blockIdx.x * 64 + wave * 16;

    // Q^T B-operands for this wave's 16 queries (kept in registers all loop).
    const _Float16* qrow = q + ((size_t)bh * NN + (qbase + col)) * DIMD;
    const half16_t Qb0 = pack16(qrow +  0 + hi * 8, qrow + 16 + hi * 8);
    const half16_t Qb1 = pack16(qrow + 32 + hi * 8, qrow + 48 + hi * 8);

    float8_t o[4] = {{}, {}, {}, {}};
    float m_run = -1e30f, l_run = 0.0f;

    const _Float16* kbh = k + (size_t)bh * NN * DIMD;
    const _Float16* vbh = v + (size_t)bh * DIMD * NN;

    // Issue the 8 async b128 copies (per wave) for one 64-key block.
    auto issue_block = [&](int kb, int buf) {
#pragma unroll
        for (int j = 0; j < 4; ++j) {
            const int c = threadIdx.x + 128 * j;          // K chunk 0..511
            async_copy_b128(kbh + (size_t)kb * DIMD + c * 8,
                            (uint32_t)(uintptr_t)&Ks[buf][c * 8]);
        }
#pragma unroll
        for (int j = 0; j < 4; ++j) {
            const int c   = threadIdx.x + 128 * j;        // V chunk 0..511
            const int row = c >> 3, off = (c & 7) * 8;    // 64 rows x 128B
            async_copy_b128(vbh + (size_t)row * NN + kb + off,
                            (uint32_t)(uintptr_t)&Vs[buf][row * BC + off]);
        }
    };

    issue_block(0, 0);                                    // prologue

    for (int kb = 0; kb < NN; kb += BC) {
        const int buf = (kb / BC) & 1;
        __syncthreads();   // all waves done reading buf^1 from 2 blocks ago
        if (kb + BC < NN) {
            issue_block(kb + BC, buf ^ 1);
            if (kb + 2 * BC < NN)   // warm L2 for the block after next
                __builtin_prefetch(kbh + (size_t)(kb + 2 * BC) * DIMD, 0, 1);
            asm volatile("s_wait_asynccnt 0x8" ::: "memory");  // block kb landed
        } else {
            asm volatile("s_wait_asynccnt 0x0" ::: "memory");
        }
        __syncthreads();   // every thread's share of block kb is in LDS

        // ---- S^T = K Q^T : four 16-key tiles, contraction over d in 2x32 ----
        float8_t s[4];
#pragma unroll
        for (int t = 0; t < 4; ++t) {
            const _Float16* kr = &Ks[buf][(size_t)(t * 16 + col) * DIMD];
            float8_t acc = {};
            acc  = wmma_f16(pack16(kr +  0 + hi * 8, kr + 16 + hi * 8), Qb0, acc);
            s[t] = wmma_f16(pack16(kr + 32 + hi * 8, kr + 48 + hi * 8), Qb1, acc);
        }

        // ---- online softmax (log2 domain: subtract + v_exp_f32 only) ----
        float mloc = -1e30f;
#pragma unroll
        for (int t = 0; t < 4; ++t)
#pragma unroll
            for (int e = 0; e < 8; ++e)
                mloc = fmaxf(mloc, s[t][e]);
        mloc = fmaxf(mloc, __shfl_xor(mloc, 16, 32));
        const float m_new = fmaxf(m_run, mloc);
        const float corr  = exp2f(m_run - m_new);

        float lsum = 0.0f;
        half16_t pb01, pb23;
#pragma unroll
        for (int e = 0; e < 8; ++e) {
            const float p0 = exp2f(s[0][e] - m_new);
            const float p1 = exp2f(s[1][e] - m_new);
            const float p2 = exp2f(s[2][e] - m_new);
            const float p3 = exp2f(s[3][e] - m_new);
            lsum += (p0 + p1) + (p2 + p3);
            pb01[e] = (_Float16)p0;  pb01[8 + e] = (_Float16)p1;
            pb23[e] = (_Float16)p2;  pb23[8 + e] = (_Float16)p3;
        }
        lsum  += __shfl_xor(lsum, 16, 32);
        l_run  = l_run * corr + lsum;
        m_run  = m_new;
#pragma unroll
        for (int g = 0; g < 4; ++g)
#pragma unroll
            for (int e = 0; e < 8; ++e) o[g][e] *= corr;

        // ---- O += V P^T : four 16-d tiles, contraction over 64 keys in 2x32 ----
#pragma unroll
        for (int g = 0; g < 4; ++g) {
            const _Float16* vr = &Vs[buf][(size_t)(g * 16 + col) * BC];
            o[g] = wmma_f16(pack16(vr +  0 + hi * 8, vr + 16 + hi * 8), pb01, o[g]);
            o[g] = wmma_f16(pack16(vr + 32 + hi * 8, vr + 48 + hi * 8), pb23, o[g]);
        }
    }

    // ---- epilogue: normalize, scatter to [B][D][N] with c = d*H + h ----
    const float inv_l = 1.0f / l_run;
    const int n = qbase + col;
#pragma unroll
    for (int g = 0; g < 4; ++g)
#pragma unroll
        for (int r = 0; r < 8; ++r) {
            const int d = g * 16 + hi * 8 + r;
            xout[((size_t)b * DD + (d * HH + h)) * NN + n] = o[g][r] * inv_l;
        }
}

// ---------------------------------------------------------------------------
// Final bottleneck conv: f32 in (attention result), f32 out, [B][D][N].
// ---------------------------------------------------------------------------
__global__ __launch_bounds__(128) void outproj_kernel(
    const float* __restrict__ x,  const float* __restrict__ W1,
    const float* __restrict__ b1, const float* __restrict__ W2,
    const float* __restrict__ b2, float* __restrict__ out)
{
    __shared__ float W1s[DBB * DD];
    __shared__ float W2s[DD * DBB];
    for (int i = threadIdx.x; i < DBB * DD; i += 128) {
        W1s[i] = W1[i];
        W2s[i] = W2[i];
    }
    __syncthreads();

    const int blk = blockIdx.x;
    const int b   = blk / (NN / 128);
    const int n   = (blk % (NN / 128)) * 128 + threadIdx.x;

    float hacc[DBB];
#pragma unroll
    for (int i = 0; i < DBB; ++i) hacc[i] = b1[i];

    const float* xb = x + (size_t)b * DD * NN + n;
    for (int ci = 0; ci < DD; ++ci) {
        float xv = xb[(size_t)ci * NN];
#pragma unroll
        for (int co = 0; co < DBB; ++co)
            hacc[co] = fmaf(W1s[co * DD + ci], xv, hacc[co]);
    }

    for (int co = 0; co < DD; ++co) {
        float oacc = b2[co];
#pragma unroll
        for (int i = 0; i < DBB; ++i)
            oacc = fmaf(W2s[co * DBB + i], hacc[i], oacc);
        out[((size_t)b * DD + co) * NN + n] = oacc;
    }
}

// ---------------------------------------------------------------------------
extern "C" void kernel_launch(void* const* d_in, const int* in_sizes, int n_in,
                              void* d_out, int out_size, void* d_ws, size_t ws_size,
                              hipStream_t stream) {
    const float* query = (const float*)d_in[0];
    const float* key   = (const float*)d_in[1];
    const float* value = (const float*)d_in[2];
    const float* Wq1 = (const float*)d_in[3];  const float* bq1 = (const float*)d_in[4];
    const float* Wq2 = (const float*)d_in[5];  const float* bq2 = (const float*)d_in[6];
    const float* Wk1 = (const float*)d_in[7];  const float* bk1 = (const float*)d_in[8];
    const float* Wk2 = (const float*)d_in[9];  const float* bk2 = (const float*)d_in[10];
    const float* Wv1 = (const float*)d_in[11]; const float* bv1 = (const float*)d_in[12];
    const float* Wv2 = (const float*)d_in[13]; const float* bv2 = (const float*)d_in[14];
    const float* Wm1 = (const float*)d_in[15]; const float* bm1 = (const float*)d_in[16];
    const float* Wm2 = (const float*)d_in[17]; const float* bm2 = (const float*)d_in[18];
    // d_in[19] = hanmingsim placeholder (unused)

    // Workspace layout: q/k/v f16 projections + f32 attention output.
    const size_t qkv_elems = (size_t)BB * HH * NN * DIMD;   // 4.19M halves each
    _Float16* q_h = (_Float16*)d_ws;
    _Float16* k_h = q_h + qkv_elems;
    _Float16* v_h = k_h + qkv_elems;
    float*    x_f = (float*)(v_h + qkv_elems);              // B*D*N floats

    const dim3 pgrid(BB * (NN / 128));
    const dim3 pblk(128);
    proj_kernel<<<pgrid, pblk, 0, stream>>>(query, Wq1, bq1, Wq2, bq2, q_h, 0, QSCALE);
    proj_kernel<<<pgrid, pblk, 0, stream>>>(key,   Wk1, bk1, Wk2, bk2, k_h, 0, 1.0f);
    proj_kernel<<<pgrid, pblk, 0, stream>>>(value, Wv1, bv1, Wv2, bv2, v_h, 2, 1.0f);

    attn_kernel<<<dim3(NN / 64, BB * HH), dim3(128), 0, stream>>>(q_h, k_h, v_h, x_f);

    outproj_kernel<<<pgrid, pblk, 0, stream>>>(x_f, Wm1, bm1, Wm2, bm2, (float*)d_out);
}
